// LayerUpdate_51556787421702
// MI455X (gfx1250) — matrix operand order repaired and checked by reading, not compiled, not run on hardware
//
#include <hip/hip_runtime.h>

// ---------------------------------------------------------------------------
// MI455X / gfx1250: bf16 WMMA implementation of the LayerUpdate pipeline.
// All matmuls (big per-batch GEMMs and per-row P=64 MLPs) run through
// v_wmma_f32_16x16x32_bf16; elementwise math stays f32.
// Big GEMM: 2 m-tiles per wave (B-fragment reuse), 64-wide K staging,
// GLOBAL_LOAD_ASYNC_TO_LDS path (ASYNCcnt) when the builtin exists.
// ---------------------------------------------------------------------------

typedef __bf16 bf16_t;
typedef __attribute__((ext_vector_type(16))) __bf16 v16bf;
typedef __attribute__((ext_vector_type(8)))  __bf16 v8bf;
typedef __attribute__((ext_vector_type(8)))  float  v8f;

static constexpr int Bc   = 32;
static constexpr int D0c  = 1024;
static constexpr int D1c  = 2048;
static constexpr int D2c  = 2048;
static constexpr int DOUTc= 10;
static constexpr int Pc   = 64;

#define DEV __device__ __forceinline__

// Optional CDNA5 async global->LDS copy path (guarded; falls back to b128 copy)
#if defined(__has_builtin)
#  if __has_builtin(__builtin_amdgcn_global_load_async_to_lds_b128) && \
      __has_builtin(__builtin_amdgcn_s_wait_asynccnt)
#    define USE_ASYNC_LDS 1
#  endif
#endif
#ifndef USE_ASYNC_LDS
#  define USE_ASYNC_LDS 0
#endif

#if USE_ASYNC_LDS
// Signature (from compiler diagnostic): arg0 = v4i* in AS1 (global), non-const.
typedef int v4i_async __attribute__((vector_size(16)));
typedef __attribute__((address_space(1))) v4i_async* as1_v4i_p;
typedef __attribute__((address_space(3))) v4i_async* as3_v4i_p;
#endif

// ---- WMMA fragment helpers (layouts per CDNA5 ISA §7.12.2) ------

DEV v16bf combine8(v8bf lo, v8bf hi) {
  v16bf r;
#pragma unroll
  for (int i = 0; i < 8; ++i) { r[i] = lo[i]; r[i + 8] = hi[i]; }
  return r;
}

// A (16xK row-major bf16 tile). row = &tile[m*stride + ks].
// lanes 0-15: K = ks+0..7 and ks+16..23 ; lanes 16-31: K = ks+8..15, ks+24..31
DEV v16bf load_a_frag(const bf16_t* row, int half) {
  const int k0 = half * 8;
  v8bf lo = *(const v8bf*)(row + k0);
  v8bf hi = *(const v8bf*)(row + k0 + 16);
  return combine8(lo, hi);
}

// B from W^T stored row-major [N][K] (== W[out][in] native). row = &W[n*K + ks]
// lanes 0-15: K = ks+0..7 and ks+16..23 ; lanes 16-31: K = ks+8..15, ks+24..31
DEV v16bf load_b_frag(const bf16_t* row, int half) {
  const int k0 = half * 8;
  v8bf lo = *(const v8bf*)(row + k0);
  v8bf hi = *(const v8bf*)(row + k0 + 16);
  return combine8(lo, hi);
}

DEV v8f wmma_bf16(v16bf a, v16bf b, v8f c) {
  return __builtin_amdgcn_wmma_f32_16x16x32_bf16(false, a, false, b,
                                                 (short)0, c, false, false);
}

// GEMM: [16 x K] (LDS slab, row-major, stride astride) x W^T[64 x K] (global)
// accumulators pre-loaded with bias.
template <int K>
DEV void mlp_gemm(const bf16_t* __restrict__ aslab, int astride,
                  const bf16_t* __restrict__ wg,   // bf16 [64][K] row-major
                  const float*  __restrict__ bias, // f32 [64]
                  v8f acc[4]) {
  const int lane = threadIdx.x & 31;
  const int half = lane >> 4;
  const int mr   = lane & 15;
#pragma unroll
  for (int nt = 0; nt < 4; ++nt) {
    const float bv = bias ? bias[nt * 16 + mr] : 0.0f;
#pragma unroll
    for (int v = 0; v < 8; ++v) acc[nt][v] = bv;
  }
#pragma unroll
  for (int ks = 0; ks < K; ks += 32) {
    v16bf a = load_a_frag(aslab + mr * astride + ks, half);
#pragma unroll
    for (int nt = 0; nt < 4; ++nt) {
      v16bf b = load_b_frag(wg + (size_t)(nt * 16 + mr) * K + ks, half);
      acc[nt] = wmma_bf16(a, b, acc[nt]);
    }
  }
}

// Store C tile (f32 accumulators, 16x64) into an LDS slab as bf16.
DEV void acc_to_slab(const v8f acc[4], bf16_t* slab, int stride, int coloff,
                     bool relu) {
  const int lane = threadIdx.x & 31;
  const int half = lane >> 4;
  const int nlo  = lane & 15;
#pragma unroll
  for (int nt = 0; nt < 4; ++nt) {
    const int n = coloff + nt * 16 + nlo;
#pragma unroll
    for (int v = 0; v < 8; ++v) {
      const int m = v + half * 8;
      float x = acc[nt][v];
      if (relu) x = fmaxf(x, 0.0f);
      slab[m * stride + n] = (bf16_t)x;
    }
  }
}

// ---- prep kernels ----------------------------------------------------------

__global__ void k_cvt(const float* __restrict__ src, bf16_t* __restrict__ dst,
                      int n) {
  int i = blockIdx.x * blockDim.x + threadIdx.x;
  if (i < n) dst[i] = (bf16_t)src[i];
}

// pad [rows][ks] f32 -> [rows][kd] bf16 (zero pad)
__global__ void k_pad_w(const float* __restrict__ src, bf16_t* __restrict__ dst,
                        int rows, int ks, int kd) {
  int i = blockIdx.x * blockDim.x + threadIdx.x;
  if (i >= rows * kd) return;
  int r = i / kd, k = i % kd;
  dst[i] = (bf16_t)(k < ks ? src[r * ks + k] : 0.0f);
}

// prop_w [32][10][2048] f32 -> [32][16][2048] bf16 (zero pad rows 10..15)
__global__ void k_pad_propw(const float* __restrict__ src,
                            bf16_t* __restrict__ dst) {
  int i = blockIdx.x * blockDim.x + threadIdx.x;
  if (i >= Bc * 16 * D2c) return;
  int k = i % D2c;
  int j = (i / D2c) % 16;
  int b = i / (16 * D2c);
  dst[i] = (bf16_t)(j < DOUTc ? src[((size_t)b * DOUTc + j) * D2c + k] : 0.0f);
}

__global__ void k_zero_flags(int* flags) {
  if (threadIdx.x < 4) flags[threadIdx.x] = 0;
}

// ---- compute_ratio (two passes: global "fix" flag, then per-element) -------

__global__ void k_ratio_flag(const float* __restrict__ lb,
                             const float* __restrict__ ub, int n, int* flag) {
  int i = blockIdx.x * blockDim.x + threadIdx.x;
  if (i >= n) return;
  float lt = fminf(lb[i], 0.0f);
  float ut = fmaxf(ub[i], 0.0f);
  float diff = ut - lt;
  if (diff == 0.0f && ut != 0.0f) atomicOr(flag, 1);
}

__global__ void k_ratio(const float* __restrict__ lb, const float* __restrict__ ub,
                        int n, const int* __restrict__ flag,
                        float* __restrict__ r0, float* __restrict__ r1,
                        float* __restrict__ beta, float* __restrict__ ambi) {
  int i = blockIdx.x * blockDim.x + threadIdx.x;
  if (i >= n) return;
  float lt = fminf(lb[i], 0.0f);
  float ut = fmaxf(ub[i], 0.0f);
  float diff = ut - lt;
  bool fix = (flag[0] == 0);
  if (diff == 0.0f && fix) diff = 1e-5f;
  float a0 = ut / diff;
  float bt = -lt * a0;
  float am = (bt > 0.0f) ? 1.0f : 0.0f;
  float a1 = (1.0f - 2.0f * (a0 * am)) * am + a0;
  r0[i] = a0; r1[i] = a1; beta[i] = bt; ambi[i] = am;
}

// ---- input embedding: [B*D0,3] -> relu -> [.,64] -> [.,64] ----------------

__global__ void k_embed(const float* __restrict__ lb0,
                        const float* __restrict__ pri,
                        const float* __restrict__ ub0,
                        const bf16_t* __restrict__ Wifp,  const float* bif,
                        const bf16_t* __restrict__ Wif1p, const float* bif1,
                        float* __restrict__ mu0_out,      // f32 [B*D0][64]
                        bf16_t* __restrict__ mu0T) {      // bf16 [B][64][D0]
  __shared__ bf16_t slab_s[8][16 * 64];
  const int wave = threadIdx.x >> 5;
  const int lane = threadIdx.x & 31;
  bf16_t* slab = slab_s[wave];
  const int base = (blockIdx.x * 8 + wave) * 16;
  // stage inp [16][32] (cols 0..2 live, rest zero), stride 64
  for (int idx = lane; idx < 16 * 32; idx += 32) {
    int m = idx >> 5, k = idx & 31;
    int r = base + m;
    float v = 0.0f;
    if (k == 0) v = lb0[r];
    else if (k == 1) v = pri[r];
    else if (k == 2) v = ub0[r];
    slab[m * 64 + k] = (bf16_t)v;
  }
  v8f acc[4];
  mlp_gemm<32>(slab, 64, Wifp, bif, acc);
  acc_to_slab(acc, slab, 64, 0, true);
  mlp_gemm<64>(slab, 64, Wif1p, bif1, acc);
  const int half = lane >> 4, nlo = lane & 15;
#pragma unroll
  for (int nt = 0; nt < 4; ++nt) {
    const int n = nt * 16 + nlo;
#pragma unroll
    for (int v = 0; v < 8; ++v) {
      const int m = v + half * 8;
      const int r = base + m;
      const float x = acc[nt][v];
      mu0_out[(size_t)r * 64 + n] = x;
      const int b = r / D0c, i = r % D0c;
      mu0T[((size_t)b * 64 + n) * D0c + i] = (bf16_t)x;
    }
  }
}

// ---- big GEMM: nb[b] = Wbf[2048][K] @ muT[b][64][K]^T -> nb bf16 [b*2048][64]
// 2 m-tiles per wave (B-fragment reuse), 64-wide K staging in LDS.

template <int K>
__global__ void k_gemm(const bf16_t* __restrict__ Wbf,  // [2048][K]
                       const bf16_t* __restrict__ muT,  // [B][64][K]
                       bf16_t* __restrict__ nb) {       // [B*2048][64]
  __shared__ bf16_t bt[64 * 64];                        // [64 n][64 k]
  const int b    = blockIdx.y;
  const int wave = threadIdx.x >> 5;
  const int lane = threadIdx.x & 31;
  const int half = lane >> 4, mr = lane & 15;
  const int mbase = (blockIdx.x * 8 + wave) * 32;       // 32 rows per wave
  const bf16_t* mt  = muT + (size_t)b * 64 * K;
  const bf16_t* aR0 = Wbf + (size_t)(mbase + mr) * K;
  const bf16_t* aR1 = Wbf + (size_t)(mbase + 16 + mr) * K;
  v8f acc0[4], acc1[4];
#pragma unroll
  for (int nt = 0; nt < 4; ++nt)
#pragma unroll
    for (int v = 0; v < 8; ++v) { acc0[nt][v] = 0.0f; acc1[nt][v] = 0.0f; }

  const int t  = threadIdx.x;
  const int sn = t >> 2;              // staging row 0..63
  const int sc = (t & 3) * 16;        // staging col 0,16,32,48

  for (int ks = 0; ks < K; ks += 64) {
    // cooperative stage of muT slice [64][64] (two b128 per thread)
#if USE_ASYNC_LDS
    __builtin_amdgcn_global_load_async_to_lds_b128(
        (as1_v4i_p)(mt + (size_t)sn * K + ks + sc),
        (as3_v4i_p)(bt + sn * 64 + sc), 0, 0);
    __builtin_amdgcn_global_load_async_to_lds_b128(
        (as1_v4i_p)(mt + (size_t)sn * K + ks + sc + 8),
        (as3_v4i_p)(bt + sn * 64 + sc + 8), 0, 0);
    __builtin_amdgcn_s_wait_asynccnt(0);
#else
    *(v8bf*)(bt + sn * 64 + sc)     = *(const v8bf*)(mt + (size_t)sn * K + ks + sc);
    *(v8bf*)(bt + sn * 64 + sc + 8) = *(const v8bf*)(mt + (size_t)sn * K + ks + sc + 8);
#endif
    __builtin_prefetch(aR0 + ks + 256, 0, 1);  // global_prefetch_b8
    __builtin_prefetch(aR1 + ks + 256, 0, 1);
    __syncthreads();
#pragma unroll
    for (int kk = 0; kk < 64; kk += 32) {
      v16bf a0 = load_a_frag(aR0 + ks + kk, half);
      v16bf a1 = load_a_frag(aR1 + ks + kk, half);
#pragma unroll
      for (int nt = 0; nt < 4; ++nt) {
        v16bf bb = load_b_frag(bt + (nt * 16 + mr) * 64 + kk, half);
        acc0[nt] = wmma_bf16(a0, bb, acc0[nt]);
        acc1[nt] = wmma_bf16(a1, bb, acc1[nt]);
      }
    }
    __syncthreads();
  }
  const int nlo = lane & 15;
#pragma unroll
  for (int nt = 0; nt < 4; ++nt) {
    const int n = nt * 16 + nlo;
#pragma unroll
    for (int v = 0; v < 8; ++v) {
      const int m = v + half * 8;
      const size_t r0i = (size_t)b * 2048 + mbase + m;
      const size_t r1i = r0i + 16;
      nb[r0i * 64 + n] = (bf16_t)acc0[nt][v];
      nb[r1i * 64 + n] = (bf16_t)acc1[nt][v];
    }
  }
}

// ---- fused relu_block: 6 MLP layers per 16-row tile, all WMMA --------------

template <int D>
__global__ void k_relublock(const bf16_t* __restrict__ nbmat,  // bf16 [N][64]
                            const float* __restrict__ r0a, const float* __restrict__ r1a,
                            const float* __restrict__ betaa, const float* __restrict__ ambia,
                            const float* __restrict__ lb, const float* __restrict__ ub,
                            const float* __restrict__ dual,       // [N][2]
                            const float* __restrict__ layer_bias, // [D]
                            const float* __restrict__ mu_in,      // f32 [N][64]
                            const bf16_t* Wfc1p,  const float* bfc1,
                            const bf16_t* Wfc11p, const float* bfc11,
                            const bf16_t* Wfc3p,  const float* bfc3,
                            const bf16_t* Wfc32p, const float* bfc32,
                            const bf16_t* Wfc4p,  const float* bfc4,
                            const bf16_t* Wfc42p, const float* bfc42,
                            float* __restrict__ mu_out,           // f32 [N][64]
                            bf16_t* __restrict__ muT) {           // bf16 [B][64][D]
  __shared__ bf16_t slab_s[8][16 * 128];
  __shared__ float  rs_s[8][3][16];
  const int wave = threadIdx.x >> 5;
  const int lane = threadIdx.x & 31;
  bf16_t* slab = slab_s[wave];
  float* rs0 = rs_s[wave][0];
  float* rs1 = rs_s[wave][1];
  float* rsa = rs_s[wave][2];
  const int half = lane >> 4, mr = lane & 15;
  const int base = (blockIdx.x * 8 + wave) * 16;

  if (lane < 16) {
    const int r = base + lane;
    rs0[lane] = r0a[r]; rs1[lane] = r1a[r]; rsa[lane] = ambia[r];
  }
  // 1) n_bounds [16][6 -> pad 32], stride 128
  for (int idx = lane; idx < 16 * 32; idx += 32) {
    const int m = idx >> 5, k = idx & 31;
    const int r = base + m;
    float v = 0.0f;
    if (k == 0) v = betaa[r];
    else if (k == 1) v = lb[r];
    else if (k == 2) v = ub[r];
    else if (k == 3) v = dual[2 * (size_t)r];
    else if (k == 4) v = dual[2 * (size_t)r + 1];
    else if (k == 5) v = layer_bias[r % D];
    slab[m * 128 + k] = (bf16_t)v;
  }
  v8f acc[4], relax[4];
  mlp_gemm<32>(slab, 128, Wfc1p, bfc1, acc);
  acc_to_slab(acc, slab, 64, 0, true);
  mlp_gemm<64>(slab, 64, Wfc11p, bfc11, relax);
#pragma unroll
  for (int nt = 0; nt < 4; ++nt)
#pragma unroll
    for (int v = 0; v < 8; ++v) relax[nt][v] *= rsa[v + half * 8];

  // 2) nb_in = [nb*r0 | nb*r1] [16][128]
  for (int idx = lane; idx < 16 * 128; idx += 32) {
    const int m = idx >> 7, c = idx & 127;
    const int k = c & 63;
    const float s = (c < 64) ? rs0[m] : rs1[m];
    const float v = (float)nbmat[(size_t)(base + m) * 64 + k] * s;
    slab[m * 128 + c] = (bf16_t)v;
  }
  mlp_gemm<128>(slab, 128, Wfc3p, bfc3, acc);
  acc_to_slab(acc, slab, 128, 0, true);  // t -> cols 0..63
  for (int idx = lane; idx < 16 * 64; idx += 32) {
    const int m = idx >> 6, k = idx & 63;
    slab[m * 128 + 64 + k] = (bf16_t)mu_in[(size_t)(base + m) * 64 + k];
  }
  mlp_gemm<128>(slab, 128, Wfc32p, bfc32, acc);  // t2

  // 3) concat(relax, t2) -> fc4 -> relu -> fc42
  acc_to_slab(relax, slab, 128, 0, false);
  acc_to_slab(acc, slab, 128, 64, false);
  mlp_gemm<128>(slab, 128, Wfc4p, bfc4, acc);
  acc_to_slab(acc, slab, 64, 0, true);
  mlp_gemm<64>(slab, 64, Wfc42p, bfc42, acc);

  // mask by (r0 != 0), write f32 + transposed bf16
  const int nlo = lane & 15;
#pragma unroll
  for (int nt = 0; nt < 4; ++nt) {
    const int n = nt * 16 + nlo;
#pragma unroll
    for (int v = 0; v < 8; ++v) {
      const int m = v + half * 8;
      const size_t r = base + m;
      const float x = acc[nt][v] * ((rs0[m] != 0.0f) ? 1.0f : 0.0f);
      mu_out[r * 64 + n] = x;
      const int b = (int)(r / D), i = (int)(r % D);
      muT[((size_t)b * 64 + n) * D + i] = (bf16_t)x;
    }
  }
}

// ---- final: per-batch prop GEMM (K=2048) + output head ---------------------

__global__ void k_final(const bf16_t* __restrict__ pwbf,  // [32][16][2048]
                        const bf16_t* __restrict__ mu2T,  // [32][64][2048]
                        const float* __restrict__ lb3, const float* __restrict__ ub3,
                        const float* __restrict__ primals0,
                        const float* __restrict__ prop_b,
                        const bf16_t* Wo1p, const float* bo1,
                        const bf16_t* Wo2p, const float* bo2,
                        const bf16_t* Wo3p, const float* bo3,
                        float* __restrict__ mu3_out) {    // f32 [320][64]
  __shared__ bf16_t slab[16 * 160];
  const int b    = blockIdx.x;
  const int lane = threadIdx.x & 31;
  const int half = lane >> 4, mr = lane & 15;
  const bf16_t* A  = pwbf + (size_t)b * 16 * D2c;
  const bf16_t* Bt = mu2T + (size_t)b * 64 * D2c;
  v8f acc[4], rel[4];
#pragma unroll
  for (int nt = 0; nt < 4; ++nt)
#pragma unroll
    for (int v = 0; v < 8; ++v) acc[nt][v] = 0.0f;
  for (int ks = 0; ks < D2c; ks += 32) {
    v16bf a = load_a_frag(A + (size_t)mr * D2c + ks, half);
#pragma unroll
    for (int nt = 0; nt < 4; ++nt) {
      v16bf bb = load_b_frag(Bt + (size_t)(nt * 16 + mr) * D2c + ks, half);
      acc[nt] = wmma_bf16(a, bb, acc[nt]);
    }
  }
  acc_to_slab(acc, slab, 160, 64, false);  // nb -> cols 64..127

  // n_bounds4 [16][4 -> pad 32] -> cols 128..159
  for (int idx = lane; idx < 16 * 32; idx += 32) {
    const int m = idx >> 5, k = idx & 31;
    float v = 0.0f;
    if (m < DOUTc) {
      const int r = b * DOUTc + m;
      if (k == 0) v = lb3[r];
      else if (k == 1) v = ub3[r];
      else if (k == 2) v = primals0[r];
      else if (k == 3) v = prop_b[r];
    }
    slab[m * 160 + 128 + k] = (bf16_t)v;
  }
  v8f tmp[4];
  mlp_gemm<32>(slab + 128, 160, Wo1p, bo1, rel);
  acc_to_slab(rel, slab, 160, 0, true);    // relax -> cols 0..63
  mlp_gemm<128>(slab, 160, Wo2p, bo2, tmp);
  acc_to_slab(tmp, slab, 160, 0, true);
  mlp_gemm<64>(slab, 160, Wo3p, bo3, tmp);

  const int nlo = lane & 15;
#pragma unroll
  for (int nt = 0; nt < 4; ++nt) {
    const int n = nt * 16 + nlo;
#pragma unroll
    for (int v = 0; v < 8; ++v) {
      const int m = v + half * 8;
      if (m < DOUTc) mu3_out[(size_t)(b * DOUTc + m) * 64 + n] = tmp[nt][v];
    }
  }
}

// ---------------------------------------------------------------------------

extern "C" void kernel_launch(void* const* d_in, const int* in_sizes, int n_in,
                              void* d_out, int out_size, void* d_ws, size_t ws_size,
                              hipStream_t stream) {
  (void)in_sizes; (void)n_in; (void)out_size; (void)ws_size;
  const float* lb0   = (const float*)d_in[0];
  const float* ub0   = (const float*)d_in[1];
  const float* lb1   = (const float*)d_in[2];
  const float* ub1   = (const float*)d_in[3];
  const float* lb2   = (const float*)d_in[4];
  const float* ub2   = (const float*)d_in[5];
  const float* lb3   = (const float*)d_in[6];
  const float* ub3   = (const float*)d_in[7];
  const float* dual0 = (const float*)d_in[8];
  const float* dual1 = (const float*)d_in[9];
  const float* primals0 = (const float*)d_in[10];
  const float* prim  = (const float*)d_in[11];
  const float* mu1   = (const float*)d_in[12];
  const float* mu2   = (const float*)d_in[13];
  const float* w1    = (const float*)d_in[14];
  const float* b1    = (const float*)d_in[15];
  const float* w2    = (const float*)d_in[16];
  const float* b2    = (const float*)d_in[17];
  const float* propw = (const float*)d_in[18];
  const float* propb = (const float*)d_in[19];
  const float* Wif   = (const float*)d_in[20];  const float* bif  = (const float*)d_in[21];
  const float* Wif1  = (const float*)d_in[22];  const float* bif1 = (const float*)d_in[23];
  const float* Wfc1  = (const float*)d_in[24];  const float* bfc1 = (const float*)d_in[25];
  const float* Wfc11 = (const float*)d_in[26];  const float* bfc11= (const float*)d_in[27];
  const float* Wfc3  = (const float*)d_in[28];  const float* bfc3 = (const float*)d_in[29];
  const float* Wfc32 = (const float*)d_in[30];  const float* bfc32= (const float*)d_in[31];
  const float* Wfc4  = (const float*)d_in[32];  const float* bfc4 = (const float*)d_in[33];
  const float* Wfc42 = (const float*)d_in[34];  const float* bfc42= (const float*)d_in[35];
  const float* Wo1   = (const float*)d_in[36];  const float* bo1  = (const float*)d_in[37];
  const float* Wo2   = (const float*)d_in[38];  const float* bo2  = (const float*)d_in[39];
  const float* Wo3   = (const float*)d_in[40];  const float* bo3  = (const float*)d_in[41];

  float* out = (float*)d_out;
  float* mu0_out = out;
  float* mu1_out = out + (size_t)Bc * D0c * Pc;
  float* mu2_out = mu1_out + (size_t)Bc * D1c * Pc;
  float* mu3_out = mu2_out + (size_t)Bc * D2c * Pc;

  // workspace carve-out (256B aligned)
  char* ws = (char*)d_ws;
  size_t off = 0;
  auto alloc = [&](size_t bytes) -> void* {
    off = (off + 255) & ~(size_t)255;
    void* p = ws + off;
    off += bytes;
    return p;
  };
  bf16_t* w1bf  = (bf16_t*)alloc((size_t)D1c * D0c * 2);
  bf16_t* w2bf  = (bf16_t*)alloc((size_t)D2c * D1c * 2);
  bf16_t* pwbf  = (bf16_t*)alloc((size_t)Bc * 16 * D2c * 2);
  bf16_t* Wifp  = (bf16_t*)alloc(64 * 32 * 2);
  bf16_t* Wif1p = (bf16_t*)alloc(64 * 64 * 2);
  bf16_t* Wfc1p = (bf16_t*)alloc(64 * 32 * 2);
  bf16_t* Wfc11p= (bf16_t*)alloc(64 * 64 * 2);
  bf16_t* Wfc3p = (bf16_t*)alloc(64 * 128 * 2);
  bf16_t* Wfc32p= (bf16_t*)alloc(64 * 128 * 2);
  bf16_t* Wfc4p = (bf16_t*)alloc(64 * 128 * 2);
  bf16_t* Wfc42p= (bf16_t*)alloc(64 * 64 * 2);
  bf16_t* Wo1p  = (bf16_t*)alloc(64 * 32 * 2);
  bf16_t* Wo2p  = (bf16_t*)alloc(64 * 128 * 2);
  bf16_t* Wo3p  = (bf16_t*)alloc(64 * 64 * 2);
  bf16_t* mu0T  = (bf16_t*)alloc((size_t)Bc * 64 * D0c * 2);
  bf16_t* mu1T  = (bf16_t*)alloc((size_t)Bc * 64 * D1c * 2);
  bf16_t* mu2T  = (bf16_t*)alloc((size_t)Bc * 64 * D2c * 2);
  bf16_t* nb1   = (bf16_t*)alloc((size_t)Bc * D1c * 64 * 2);
  bf16_t* nb2   = (bf16_t*)alloc((size_t)Bc * D2c * 64 * 2);
  const int N1 = Bc * D1c, N2 = Bc * D2c;
  float* r0_1 = (float*)alloc((size_t)N1 * 4 * 4);
  float* r1_1 = r0_1 + N1; float* be_1 = r0_1 + 2 * N1; float* am_1 = r0_1 + 3 * N1;
  float* r0_2 = (float*)alloc((size_t)N2 * 4 * 4);
  float* r1_2 = r0_2 + N2; float* be_2 = r0_2 + 2 * N2; float* am_2 = r0_2 + 3 * N2;
  int* flags = (int*)alloc(4 * sizeof(int));

  const int T = 256;
  auto blks = [](size_t n, int t) { return (unsigned)((n + t - 1) / t); };

  // prep
  k_zero_flags<<<1, 32, 0, stream>>>(flags);
  k_cvt<<<blks((size_t)D1c * D0c, T), T, 0, stream>>>(w1, w1bf, D1c * D0c);
  k_cvt<<<blks((size_t)D2c * D1c, T), T, 0, stream>>>(w2, w2bf, D2c * D1c);
  k_pad_propw<<<blks((size_t)Bc * 16 * D2c, T), T, 0, stream>>>(propw, pwbf);
  k_pad_w<<<blks(64 * 32, T), T, 0, stream>>>(Wif,   Wifp,   64, 3,   32);
  k_pad_w<<<blks(64 * 64, T), T, 0, stream>>>(Wif1,  Wif1p,  64, 64,  64);
  k_pad_w<<<blks(64 * 32, T), T, 0, stream>>>(Wfc1,  Wfc1p,  64, 6,   32);
  k_pad_w<<<blks(64 * 64, T), T, 0, stream>>>(Wfc11, Wfc11p, 64, 64,  64);
  k_pad_w<<<blks(64 * 128, T), T, 0, stream>>>(Wfc3,  Wfc3p,  64, 128, 128);
  k_pad_w<<<blks(64 * 128, T), T, 0, stream>>>(Wfc32, Wfc32p, 64, 128, 128);
  k_pad_w<<<blks(64 * 128, T), T, 0, stream>>>(Wfc4,  Wfc4p,  64, 128, 128);
  k_pad_w<<<blks(64 * 64, T), T, 0, stream>>>(Wfc42, Wfc42p, 64, 64,  64);
  k_pad_w<<<blks(64 * 32, T), T, 0, stream>>>(Wo1,   Wo1p,   64, 4,   32);
  k_pad_w<<<blks(64 * 128, T), T, 0, stream>>>(Wo2,   Wo2p,   64, 128, 128);
  k_pad_w<<<blks(64 * 64, T), T, 0, stream>>>(Wo3,   Wo3p,   64, 64,  64);

  // embedding: B*D0 rows / 16 per wave / 8 waves per block
  k_embed<<<(Bc * D0c) / 128, T, 0, stream>>>(lb0, prim, ub0, Wifp, bif,
                                              Wif1p, bif1, mu0_out, mu0T);
  // layer 1 GEMM + relu_block (each block covers 8 waves * 32 rows = 256 rows)
  k_gemm<D0c><<<dim3(D1c / 256, Bc), T, 0, stream>>>(w1bf, mu0T, nb1);
  k_ratio_flag<<<blks(N1, T), T, 0, stream>>>(lb1, ub1, N1, flags + 0);
  k_ratio<<<blks(N1, T), T, 0, stream>>>(lb1, ub1, N1, flags + 0,
                                         r0_1, r1_1, be_1, am_1);
  k_relublock<D1c><<<N1 / 128, T, 0, stream>>>(
      nb1, r0_1, r1_1, be_1, am_1, lb1, ub1, dual0, b1, mu1,
      Wfc1p, bfc1, Wfc11p, bfc11, Wfc3p, bfc3, Wfc32p, bfc32,
      Wfc4p, bfc4, Wfc42p, bfc42, mu1_out, mu1T);
  // layer 2 GEMM + relu_block
  k_gemm<D1c><<<dim3(D2c / 256, Bc), T, 0, stream>>>(w2bf, mu1T, nb2);
  k_ratio_flag<<<blks(N2, T), T, 0, stream>>>(lb2, ub2, N2, flags + 1);
  k_ratio<<<blks(N2, T), T, 0, stream>>>(lb2, ub2, N2, flags + 1,
                                         r0_2, r1_2, be_2, am_2);
  k_relublock<D2c><<<N2 / 128, T, 0, stream>>>(
      nb2, r0_2, r1_2, be_2, am_2, lb2, ub2, dual1, b2, mu2,
      Wfc1p, bfc1, Wfc11p, bfc11, Wfc3p, bfc3, Wfc32p, bfc32,
      Wfc4p, bfc4, Wfc42p, bfc42, mu2_out, mu2T);
  // final prop + output head (one wave per batch sample)
  k_final<<<Bc, 32, 0, stream>>>(pwbf, mu2T, lb3, ub3, primals0, propb,
                                 Wo1p, bo1, Wo2p, bo2, Wo3p, bo3, mu3_out);
}
